// PidgeonNetLoss_15281493639904
// MI455X (gfx1250) — matrix operand order repaired and checked by reading, not compiled, not run on hardware
//
#include <hip/hip_runtime.h>

// ---------------------------------------------------------------------------
// PidgeonNetLoss (discriminative segmentation loss) for MI455X / gfx1250.
//
// Roofline: inputs are 136 MiB (y: 128 MiB f32, c: 8 MiB i32) -> ~5.8us at
// 23.3 TB/s HBM. Memory-bound single-pass segmented reduction.
// Segment-sum expressed as onehot x Y matmul via V_WMMA_F32_16X16X32_BF16
// (32 pixels / instruction, f32 accumulate). One-hot built branch-free:
//   ((labpair ^ m*0x10001) -pk16 1) & 0x3F803F80   -> {0, bf16(1.0)} pairs
// = 3 VALU per packed pair (v_xor + v_pk_sub_u16 + v_and), no v_cmp chains.
//
// NOTE: wave-pairing (splitting label halves across waves) was tried and
// reverted: one-hot work is invariant (pixels x labels), so it only added
// partner-tile LDS re-reads and 2 block barriers per iteration. Waves stay
// fully independent here; B is shared by the two label-half WMMAs.
// ---------------------------------------------------------------------------

#define K_LABELS 32
#define D_CH     16
#define HW       (512 * 512)
#define NB       8
#define TPS      64                      // blocks (tiles) per sample
#define CHUNK    (HW / TPS)              // 4096 pixels per block
#define WAVES    8
#define WAVE_PIX (CHUNK / WAVES)         // 512 pixels per wave
#define ITER_PIX 128                     // pixels per wave iteration
#define ITERS    (WAVE_PIX / ITER_PIX)   // 4
#define YSTRIDE  68                      // padded row stride (u32 bf16-pairs), 16B aligned
#define WS_ENT   18                      // per (b,k): 16 sums + count + sumsq

typedef __attribute__((ext_vector_type(16))) __bf16         v16bf;
typedef __attribute__((ext_vector_type(2)))  __bf16         v2bf;
typedef __attribute__((ext_vector_type(8)))  float          v8f;
typedef __attribute__((ext_vector_type(8)))  unsigned int   v8u;
typedef __attribute__((ext_vector_type(2)))  unsigned short v2u16;

// f32 -> packed bf16 pair via hardware cvt (v_cvt_pk_bf16_f32)
__device__ __forceinline__ unsigned bf16pk(float a, float b) {
  v2bf p = {(__bf16)a, (__bf16)b};
  return __builtin_bit_cast(unsigned, p);
}

__global__ void seg_zero_15281493639904(float* ws, float* out) {
  int i = blockIdx.x * 256 + threadIdx.x;
  if (i < NB * K_LABELS * WS_ENT) ws[i] = 0.0f;
  if (i == 0) out[0] = 0.0f;
}

__global__ __launch_bounds__(256) void seg_partial_15281493639904(
    const float* __restrict__ y, const int* __restrict__ c,
    float* __restrict__ ws) {
  // wave-private staging tiles + block-shared partial accumulators
  __shared__ unsigned ytile[WAVES][D_CH][YSTRIDE];  // bf16 pixel-pairs, channel rows
  __shared__ unsigned lab2[WAVES][ITER_PIX / 2];    // packed u16 label pairs
  __shared__ float    bsum[K_LABELS * WS_ENT];

  const int tid  = threadIdx.x;
  const int wave = tid >> 5;
  const int lane = tid & 31;
  const int b    = blockIdx.y;

  for (int i = tid; i < K_LABELS * WS_ENT; i += 256) bsum[i] = 0.0f;
  __syncthreads();

  const float* yb = y + (size_t)b * D_CH * HW;
  const int*   cb = c + (size_t)b * HW;
  const int pixbase0 = blockIdx.x * CHUNK + wave * WAVE_PIX;

  v8f accS0 = {};  // sums for labels 0..15  (M=label, N=channel)
  v8f accS1 = {};  // sums for labels 16..31

  const int      ch   = lane & 15;
  const int      half = lane >> 4;
  const unsigned mm0  = (unsigned)ch * 0x00010001u;  // {m, m} packed u16

  for (int it = 0; it < ITERS; ++it) {
    const int p4 = pixbase0 + it * ITER_PIX + 4 * lane;

    // ---- stage 128 pixels: packed labels + bf16-packed y, per-pixel |y|^2
    int4 labs = *(const int4*)(cb + p4);
    uint2 lpk = make_uint2((unsigned)labs.x | ((unsigned)labs.y << 16),
                           (unsigned)labs.z | ((unsigned)labs.w << 16));
    *(uint2*)&lab2[wave][2 * lane] = lpk;            // ds_store_b64

    float s0 = 0.f, s1 = 0.f, s2 = 0.f, s3 = 0.f;
#pragma unroll
    for (int d = 0; d < D_CH; ++d) {
      float4 v = *(const float4*)(yb + (size_t)d * HW + p4);
      s0 = fmaf(v.x, v.x, s0);
      s1 = fmaf(v.y, v.y, s1);
      s2 = fmaf(v.z, v.z, s2);
      s3 = fmaf(v.w, v.w, s3);
      uint2 pk = make_uint2(bf16pk(v.x, v.y), bf16pk(v.z, v.w));
      *(uint2*)&ytile[wave][d][2 * lane] = pk;       // ds_store_b64
    }

    // exact f32 counts & sum-of-squares via LDS float atomics
    atomicAdd(&bsum[labs.x * WS_ENT + 16], 1.0f);
    atomicAdd(&bsum[labs.x * WS_ENT + 17], s0);
    atomicAdd(&bsum[labs.y * WS_ENT + 16], 1.0f);
    atomicAdd(&bsum[labs.y * WS_ENT + 17], s1);
    atomicAdd(&bsum[labs.z * WS_ENT + 16], 1.0f);
    atomicAdd(&bsum[labs.z * WS_ENT + 17], s2);
    atomicAdd(&bsum[labs.w * WS_ENT + 16], 1.0f);
    atomicAdd(&bsum[labs.w * WS_ENT + 17], s3);

    // ---- 4 x (32 pixels): sums[K,D] += onehot[K,32px] x Y[32px,D]
#pragma unroll
    for (int sub = 0; sub < 4; ++sub) {
      const int bp = sub * 16;  // pixel-pair base for this 32-pixel group

      // B (32x16 bf16), k order chosen here (same as A): VGPR v, lanes0-15 hold K={2v,2v+1}, lanes16-31 K=+16
      // -> lane (ch,half) reads 8 consecutive pairs at row ch.
      uint4 t0 = *(const uint4*)&ytile[wave][ch][bp + 8 * half];
      uint4 t1 = *(const uint4*)&ytile[wave][ch][bp + 8 * half + 4];
      v8u braw = {t0.x, t0.y, t0.z, t0.w, t1.x, t1.y, t1.z, t1.w};
      v16bf B = __builtin_bit_cast(v16bf, braw);

      // A (16x32 bf16 onehot), same k order as B: VGPR j holds K-pair kp=j+8*half,
      // lanes 0-15 carry M=0..15 (label row). Packed label pairs line up with
      // A's packed-pair layout -> 2 broadcast b128 loads feed the xor directly.
      uint4 lq0 = *(const uint4*)&lab2[wave][bp + 8 * half];
      uint4 lq1 = *(const uint4*)&lab2[wave][bp + 8 * half + 4];
      unsigned lp[8] = {lq0.x, lq0.y, lq0.z, lq0.w, lq1.x, lq1.y, lq1.z, lq1.w};

      v8u a0raw, a1raw;
#pragma unroll
      for (int j = 0; j < 8; ++j) {
        unsigned x0 = lp[j] ^ mm0;            // halves: lab^m  (0 iff equal)
        unsigned x1 = x0 ^ 0x00100010u;       // halves: lab^(m+16)
        v2u16 q0 = __builtin_bit_cast(v2u16, x0) - (unsigned short)1;
        v2u16 q1 = __builtin_bit_cast(v2u16, x1) - (unsigned short)1;
        a0raw[j] = __builtin_bit_cast(unsigned, q0) & 0x3F803F80u;
        a1raw[j] = __builtin_bit_cast(unsigned, q1) & 0x3F803F80u;
      }
      v16bf A0 = __builtin_bit_cast(v16bf, a0raw);
      v16bf A1 = __builtin_bit_cast(v16bf, a1raw);

      accS0 = __builtin_amdgcn_wmma_f32_16x16x32_bf16(
          false, A0, false, B, (short)0, accS0, false, false);
      accS1 = __builtin_amdgcn_wmma_f32_16x16x32_bf16(
          false, A1, false, B, (short)0, accS1, false, false);
    }
  }

  // ---- drain WMMA accumulators (C/D: reg r, lane<16 -> M=r; lane>=16 -> M=r+8)
#pragma unroll
  for (int r = 0; r < 8; ++r) {
    atomicAdd(&bsum[(r + 8 * half) * WS_ENT + ch],      accS0[r]);
    atomicAdd(&bsum[(16 + r + 8 * half) * WS_ENT + ch], accS1[r]);
  }
  __syncthreads();

  // ---- one global flush per block
  float* wsb = ws + (size_t)b * K_LABELS * WS_ENT;
  for (int i = tid; i < K_LABELS * WS_ENT; i += 256) atomicAdd(&wsb[i], bsum[i]);
}

__global__ __launch_bounds__(256) void seg_finalize_15281493639904(
    const float* __restrict__ ws, float* __restrict__ out) {
  __shared__ float cent[NB][K_LABELS][D_CH];
  __shared__ float red[256];

  const int t = threadIdx.x;
  const int b = t >> 5;
  const int k = t & 31;
  const float* e = ws + ((size_t)b * K_LABELS + k) * WS_ENT;

  float cnt = e[16];
  float sqy = e[17];
  float cd[D_CH];
  float csq = 0.f;
#pragma unroll
  for (int d = 0; d < D_CH; ++d) {
    cd[d] = e[d] / cnt;
    csq = fmaf(cd[d], cd[d], csq);
    cent[b][k][d] = cd[d];
  }
  float sq  = fmaxf(sqy - cnt * csq, 0.0f);
  float fro = sqrtf(sq + 1e-12f);
  float hv  = fmaxf(fro - 1.0f, 0.0f);       // VAR_MARGIN = 1.0
  float var = hv * hv;

  __syncthreads();

  const int kp = (k + K_LABELS - 1) & (K_LABELS - 1);  // roll by 1
  float dsq = 0.f;
#pragma unroll
  for (int d = 0; d < D_CH; ++d) {
    float dd = cd[d] - cent[b][kp][d];
    dsq = fmaf(dd, dd, dsq);
  }
  float dist = sqrtf(dsq + 1e-12f);
  float hd   = fmaxf(3.0f - dist, 0.0f);     // DIST_MARGIN = 3.0
  float dis  = hd * hd;

  // loss = mean_b(mean_k(var)) + mean_b(mean_k(dis)) (ALPHA=BETA=1)
  red[t] = (var + dis) * (1.0f / (NB * K_LABELS));
  __syncthreads();
  for (int s = 128; s > 0; s >>= 1) {
    if (t < s) red[t] += red[t + s];
    __syncthreads();
  }
  if (t == 0) out[0] = red[0];
}

extern "C" void kernel_launch(void* const* d_in, const int* in_sizes, int n_in,
                              void* d_out, int out_size, void* d_ws,
                              size_t ws_size, hipStream_t stream) {
  (void)in_sizes; (void)n_in; (void)out_size; (void)ws_size;
  const float* y  = (const float*)d_in[0];
  const int*   c  = (const int*)d_in[1];
  float*       out = (float*)d_out;
  float*       ws  = (float*)d_ws;

  seg_zero_15281493639904<<<(NB * K_LABELS * WS_ENT + 255) / 256, 256, 0,
                            stream>>>(ws, out);
  dim3 grid(TPS, NB);
  seg_partial_15281493639904<<<grid, 256, 0, stream>>>(y, c, ws);
  seg_finalize_15281493639904<<<1, 256, 0, stream>>>(ws, out);
}
